// TLdecoder_62809601736759
// MI455X (gfx1250) — matrix operand order repaired and checked by reading, not compiled, not run on hardware
//
#include <hip/hip_runtime.h>
#include <hip/hip_bf16.h>
#include <math.h>

// ---------------------------------------------------------------------------
// TL-decoder for MI455X (gfx1250).
// Single-key softmax == 1 collapses two of the three attentions to plain
// GEMM chains, and kills the scan dependency: all 1025 steps are independent.
// Batched rows R = 16*1025 = 16400. All GEMMs via v_wmma_f32_16x16x32_bf16.
// A-tiles are staged to LDS with the Tensor Data Mover (tensor_load_to_lds,
// TENSORcnt) so the 8 waves of a block share one DMA instead of 8x redundant
// global loads.
// ---------------------------------------------------------------------------

typedef __bf16 bf16_t;
typedef __attribute__((ext_vector_type(16))) __bf16 v16bf;
typedef __attribute__((ext_vector_type(8)))  __bf16 v8bf;
typedef __attribute__((ext_vector_type(8)))  float  v8f;
typedef unsigned int u32x4 __attribute__((ext_vector_type(4)));
typedef int          i32x4 __attribute__((ext_vector_type(4)));
typedef int          i32x8 __attribute__((ext_vector_type(8)));

#define D_MODEL 512
#define SEQ     1024
#define BS      16
#define LSTEP   1025              // SEQ + 1 (h0 prepended)
#define R_ROWS  (BS * LSTEP)      // 16400
#define R_PAD   16480             // padded rows so tile-65 A-frag loads stay in-bounds
#define RK_ROWS (BS * SEQ)        // 16384
#define D_FF    2048
#define ATT_SCALE 0.125f          // 1/sqrt(64)
#define LN_EPS  1e-5f

__device__ __forceinline__ v8f wmma_bf16(v16bf a, v16bf b, v8f c) {
  return __builtin_amdgcn_wmma_f32_16x16x32_bf16(false, a, false, b, (short)0, c,
                                                 false, false);
}

// A fragment (16x32 bf16): lane L -> row = L&15; lo lanes K {0..15},
// hi lanes K {16..31} (two 16B chunks).
__device__ __forceinline__ v16bf load_a_frag(const bf16_t* rowp, int k0, int hi) {
  const bf16_t* p = rowp + k0 + hi * 16;
  v8bf c0 = *(const v8bf*)p;
  v8bf c1 = *(const v8bf*)(p + 8);
  v16bf a;
#pragma unroll
  for (int i = 0; i < 8; ++i) { a[i] = c0[i]; a[8 + i] = c1[i]; }
  return a;
}

// ---------------------------------------------------------------------------
// Tensor Data Mover: DMA a 2D tile (rows x width bf16, row stride in elems)
// from global memory into LDS. D# built per CDNA5 ISA §8.3/8.4:
//  group0: count=1 | lds_addr | global_addr[56:0] | type=2
//  group1: data_size=2B | tensor_dim0/1 | tile_dim0/1 | tensor_dim0_stride
//  groups 2/3 (and trailing group on this toolchain's 6-arg builtin): zero.
// ---------------------------------------------------------------------------
__device__ __forceinline__ void tdm_load_tile_2d(unsigned int lds_off,
                                                 const bf16_t* g, unsigned int width,
                                                 unsigned int rows,
                                                 unsigned int stride) {
  unsigned long long ga = (unsigned long long)(uintptr_t)g;
  u32x4 g0;
  g0.x = 1u;                                              // count = 1 (valid)
  g0.y = lds_off;                                         // D#.lds_addr
  g0.z = (unsigned int)ga;                                // global_addr[31:0]
  g0.w = (unsigned int)((ga >> 32) & 0x01FFFFFFu) | (2u << 30);  // [56:32] | type=2
  i32x8 g1;
  g1[0] = (int)(1u << 16);                                // data_size code 1 (2 bytes)
  g1[1] = (int)((width & 0xFFFFu) << 16);                 // tensor_dim0[15:0]
  g1[2] = (int)((width >> 16) | ((rows & 0xFFFFu) << 16)); // dim0 hi | tensor_dim1 lo
  g1[3] = (int)((rows >> 16) | (width << 16));            // dim1 hi | tile_dim0
  g1[4] = (int)(rows & 0xFFFFu);                          // tile_dim1 (tile_dim2=0)
  g1[5] = (int)stride;                                    // tensor_dim0_stride lo32
  g1[6] = 0;
  g1[7] = 0;
  i32x4 z4 = {0, 0, 0, 0};
  i32x8 z8 = {0, 0, 0, 0, 0, 0, 0, 0};
  __builtin_amdgcn_tensor_load_to_lds(g0, g1, z4, z4, z8, 0);
}

// ---------------------------------------------------------------------------
// Generic WMMA GEMM:  out[r, n] = sum_k A[r,k] * BT[n,k]   (i.e. act @ W.T)
// Block = 256 threads = 8 waves; each wave owns a 16x64 output tile.
// The 16xK A tile is TDM-staged to LDS once per block and shared by all waves.
// ---------------------------------------------------------------------------
enum { EPI_BF16 = 0, EPI_F32 = 1, EPI_GELU_BF16 = 2, EPI_BIAS_F32 = 3, EPI_VGT = 4 };

template <int EPI, int KSZ>
__global__ __launch_bounds__(256) void gemm_kernel(
    const bf16_t* __restrict__ A, int lda, const bf16_t* __restrict__ BT, int ldb,
    float* __restrict__ outF, bf16_t* __restrict__ outB, int ldo,
    const float* __restrict__ bias) {
  __shared__ __align__(16) bf16_t ldsA[16 * KSZ];
  const int lane = threadIdx.x & 31;
  const int wid  = threadIdx.x >> 5;
  const int m0   = blockIdx.x * 16;
  const int nb   = blockIdx.y * 512 + wid * 64;
  const int mrow = lane & 15;
  const int hi   = lane >> 4;

  if (threadIdx.x < 32) {  // wave 0 issues the DMA for the whole block
    tdm_load_tile_2d((unsigned int)(uintptr_t)(void*)ldsA,
                     A + (size_t)m0 * lda, (unsigned int)KSZ, 16u,
                     (unsigned int)lda);
    __builtin_amdgcn_s_wait_tensorcnt(0);
  }
  __syncthreads();

  const bf16_t* arow = ldsA + (size_t)mrow * KSZ;
  v8f acc[4] = {};
  for (int k0 = 0; k0 < KSZ; k0 += 32) {
    v16bf a = load_a_frag(arow, k0, hi);  // ds_load_b128 x2 from the TDM-staged tile
#pragma unroll
    for (int j = 0; j < 4; ++j) {
      // B fragment: lane -> column n = nb+j*16+(L&15); 16 contiguous K values.
      const bf16_t* bp = BT + (size_t)(nb + j * 16 + mrow) * ldb + k0 + hi * 16;
      v16bf bfv = *(const v16bf*)bp;
      acc[j] = wmma_bf16(a, bfv, acc[j]);
    }
  }
  // C layout: VGPR v -> row v (lanes 0-15) / v+8 (lanes 16-31), col = L&15.
#pragma unroll
  for (int j = 0; j < 4; ++j) {
    int n = nb + j * 16 + mrow;
#pragma unroll
    for (int v = 0; v < 8; ++v) {
      int row = m0 + v + hi * 8;
      float x = acc[j][v];
      if (EPI == EPI_BF16) {
        outB[(size_t)row * ldo + n] = (bf16_t)x;
      } else if (EPI == EPI_F32) {
        outF[(size_t)row * ldo + n] = x;
      } else if (EPI == EPI_GELU_BF16) {
        x += bias[n];
        x = 0.5f * x * (1.0f + erff(x * 0.70710678118654752f));
        outB[(size_t)row * ldo + n] = (bf16_t)x;
      } else if (EPI == EPI_BIAS_F32) {
        outF[(size_t)row * ldo + n] = x + bias[n];
      } else {  // EPI_VGT: store Vg transposed as [b][c=h*64+d][n] for P@V B-frags
        int b = row >> 10, nin = row & 1023;
        outB[((size_t)(b * 512 + n)) * 1024 + nin] = (bf16_t)x;
      }
    }
  }
}

// ---------------------------------------------------------------------------
// Flash attention: per (b,h,qtile of 16 rows), 1025 queries vs 1024 keys.
// S via WMMA (K=d), online softmax on C-fragments, P bounced through LDS to
// become an A-fragment, O += P@V via WMMA against transposed Vg.
// ---------------------------------------------------------------------------
__global__ __launch_bounds__(128) void attn_kernel(
    const bf16_t* __restrict__ Q, const bf16_t* __restrict__ Kg,
    const bf16_t* __restrict__ VgT, bf16_t* __restrict__ att) {
  __shared__ bf16_t ldsP[4][16 * 32];
  const int lane = threadIdx.x & 31;
  const int wid  = threadIdx.x >> 5;
  const int bh   = blockIdx.y;
  const int b    = bh >> 3;
  const int h    = bh & 7;
  const int qt   = blockIdx.x * 4 + wid;
  if (qt >= 65) return;  // 65 query tiles of 16 cover 1025 steps (last has 1 row)
  const int mrow = lane & 15;
  const int hi   = lane >> 4;

  // Q A-fragments with the softmax scale folded in.
  v16bf qa[2];
  {
    const bf16_t* qr = Q + ((size_t)(b * LSTEP + qt * 16 + mrow)) * 512 + h * 64;
#pragma unroll
    for (int f = 0; f < 2; ++f) {
      const bf16_t* p = qr + f * 32 + hi * 16;
      v8bf c0 = *(const v8bf*)p;
      v8bf c1 = *(const v8bf*)(p + 8);
      v16bf a;
#pragma unroll
      for (int i = 0; i < 8; ++i) {
        a[i]     = (bf16_t)((float)c0[i] * ATT_SCALE);
        a[8 + i] = (bf16_t)((float)c1[i] * ATT_SCALE);
      }
      qa[f] = a;
    }
  }

  v8f O0 = {}, O1 = {}, O2 = {}, O3 = {};
  float rmax[8], lsum[8];
#pragma unroll
  for (int v = 0; v < 8; ++v) { rmax[v] = -3.0e38f; lsum[v] = 0.0f; }

  bf16_t* P = ldsP[wid];
  const bf16_t* kgb = Kg + (size_t)b * 1024 * 512 + h * 64;
  const bf16_t* vgb = VgT + ((size_t)(b * 512 + h * 64)) * 1024;

  for (int n0 = 0; n0 < 1024; n0 += 32) {
    // Streaming lookahead for the L2-resident K/V (global_prefetch_b8).
    if (((n0 & 127) == 0) && (n0 + 128 < 1024)) {
      __builtin_prefetch(kgb + (size_t)(n0 + 128 + mrow) * 512 + hi * 16, 0, 0);
      __builtin_prefetch(vgb + (size_t)mrow * 1024 + n0 + 128 + hi * 16, 0, 0);
    }
    v8f s0 = {}, s1 = {};
#pragma unroll
    for (int sub = 0; sub < 2; ++sub) {
      const bf16_t* kp = kgb + (size_t)(n0 + sub * 16 + mrow) * 512 + hi * 16;
      v16bf k0 = *(const v16bf*)kp;        // d = 0..31
      v16bf k1 = *(const v16bf*)(kp + 32); // d = 32..63
      if (sub == 0) { s0 = wmma_bf16(qa[0], k0, s0); s0 = wmma_bf16(qa[1], k1, s0); }
      else          { s1 = wmma_bf16(qa[0], k0, s1); s1 = wmma_bf16(qa[1], k1, s1); }
    }
    // Online softmax across the 32 new columns (16-lane butterfly per row).
#pragma unroll
    for (int v = 0; v < 8; ++v) {
      float loc = fmaxf(s0[v], s1[v]);
#pragma unroll
      for (int m = 8; m >= 1; m >>= 1) loc = fmaxf(loc, __shfl_xor(loc, m, 32));
      float nm  = fmaxf(rmax[v], loc);
      float fac = __expf(rmax[v] - nm);
      rmax[v]   = nm;
      float p0 = __expf(s0[v] - nm);
      float p1 = __expf(s1[v] - nm);
      lsum[v] = lsum[v] * fac + p0 + p1;
      O0[v] *= fac; O1[v] *= fac; O2[v] *= fac; O3[v] *= fac;
      int row = v + (hi << 3);
      P[row * 32 + mrow]      = (bf16_t)p0;
      P[row * 32 + 16 + mrow] = (bf16_t)p1;
    }
    asm volatile("s_wait_dscnt 0" ::: "memory");  // cross-lane LDS visibility (same wave)
    // Reload P as an A-fragment (16x32, K = local key index).
    v16bf pa;
    {
      const bf16_t* pp = P + mrow * 32 + hi * 16;
      v8bf c0 = *(const v8bf*)pp;
      v8bf c1 = *(const v8bf*)(pp + 8);
#pragma unroll
      for (int i = 0; i < 8; ++i) { pa[i] = c0[i]; pa[8 + i] = c1[i]; }
    }
#pragma unroll
    for (int j = 0; j < 4; ++j) {
      const bf16_t* vp = vgb + (size_t)(j * 16 + mrow) * 1024 + n0 + hi * 16;
      v16bf vf = *(const v16bf*)vp;
      if (j == 0)      O0 = wmma_bf16(pa, vf, O0);
      else if (j == 1) O1 = wmma_bf16(pa, vf, O1);
      else if (j == 2) O2 = wmma_bf16(pa, vf, O2);
      else             O3 = wmma_bf16(pa, vf, O3);
    }
  }
  // Normalize by full row sum and store merged-head output.
  float inv[8];
#pragma unroll
  for (int v = 0; v < 8; ++v) {
    float s = lsum[v];
#pragma unroll
    for (int m = 8; m >= 1; m >>= 1) s += __shfl_xor(s, m, 32);
    inv[v] = 1.0f / s;
  }
#pragma unroll
  for (int v = 0; v < 8; ++v) {
    int t = qt * 16 + v + (hi << 3);
    if (t <= SEQ) {
      size_t o = ((size_t)(b * LSTEP + t)) * 512 + h * 64 + mrow;
      att[o + 0]  = (bf16_t)(O0[v] * inv[v]);
      att[o + 16] = (bf16_t)(O1[v] * inv[v]);
      att[o + 32] = (bf16_t)(O2[v] * inv[v]);
      att[o + 48] = (bf16_t)(O3[v] * inv[v]);
    }
  }
}

// ---------------------------------------------------------------------------
// Row kernels: conv / layernorm / residual plumbing. One block per row.
// ---------------------------------------------------------------------------
__global__ __launch_bounds__(256) void cvt_kernel(const float* __restrict__ s,
                                                  bf16_t* __restrict__ d, size_t n) {
  size_t i = (size_t)blockIdx.x * 256 + threadIdx.x;
  if (i < n) d[i] = (bf16_t)s[i];
}

__global__ __launch_bounds__(256) void conv1_kernel(
    const float* __restrict__ input, const float* __restrict__ h0,
    const float* __restrict__ w, bf16_t* __restrict__ xcB) {
  __shared__ float row[514];
  int r = blockIdx.x, b = r / LSTEP, t = r % LSTEP;
  const float* src = (t == 0) ? h0 + (size_t)b * 512
                              : input + ((size_t)b * SEQ + (t - 1)) * 512;
  int tid = threadIdx.x;
  for (int i = tid; i < 512; i += 256) row[i + 1] = src[i];
  if (tid == 0) { row[0] = 0.0f; row[513] = 0.0f; }
  __syncthreads();
  float w0 = w[0], w1 = w[1], w2 = w[2];
  for (int i = tid; i < 512; i += 256)
    xcB[(size_t)r * 512 + i] = (bf16_t)(w0 * row[i] + w1 * row[i + 1] + w2 * row[i + 2]);
}

__device__ __forceinline__ float block_reduce_sum(float part, float* rbuf, int tid,
                                                  int slot) {
#pragma unroll
  for (int off = 16; off; off >>= 1) part += __shfl_down(part, off, 32);
  if ((tid & 31) == 0) rbuf[tid >> 5] = part;
  __syncthreads();
  if (tid == 0) {
    float s = 0.f;
#pragma unroll
    for (int i = 0; i < 8; ++i) s += rbuf[i];
    rbuf[slot] = s;
  }
  __syncthreads();
  return rbuf[slot];
}

// LN1: x1 = LN(gem + Xrow)*g+b ; also x2 = conv(x1, conv2_w) in bf16.
__global__ __launch_bounds__(256) void ln1_kernel(
    const float* __restrict__ gem, const float* __restrict__ input,
    const float* __restrict__ h0, const float* __restrict__ g,
    const float* __restrict__ bta, const float* __restrict__ wconv,
    float* __restrict__ x1F, bf16_t* __restrict__ x2B) {
  __shared__ float rowb[514];
  __shared__ float rbuf[12];
  int r = blockIdx.x, b = r / LSTEP, t = r % LSTEP;
  const float* res = (t == 0) ? h0 + (size_t)b * 512
                              : input + ((size_t)b * SEQ + (t - 1)) * 512;
  int tid = threadIdx.x;
  size_t base = (size_t)r * 512;
  float x0 = gem[base + tid] + res[tid];
  float x1 = gem[base + tid + 256] + res[tid + 256];
  float mean = block_reduce_sum(x0 + x1, rbuf, tid, 8) * (1.0f / 512.0f);
  float d0 = x0 - mean, d1 = x1 - mean;
  float var = block_reduce_sum(d0 * d0 + d1 * d1, rbuf, tid, 9) * (1.0f / 512.0f);
  float rstd = rsqrtf(var + LN_EPS);
  float y0 = d0 * rstd * g[tid] + bta[tid];
  float y1 = d1 * rstd * g[tid + 256] + bta[tid + 256];
  x1F[base + tid] = y0;
  x1F[base + tid + 256] = y1;
  rowb[tid + 1] = y0;
  rowb[tid + 257] = y1;
  if (tid == 0) { rowb[0] = 0.0f; rowb[513] = 0.0f; }
  __syncthreads();
  float w0 = wconv[0], w1 = wconv[1], w2 = wconv[2];
  for (int i = tid; i < 512; i += 256)
    x2B[base + i] = (bf16_t)(w0 * rowb[i] + w1 * rowb[i + 1] + w2 * rowb[i + 2]);
}

// LN2/LN3: out = LN(gem + res)*g+b, stored f32 (next residual) and bf16 (next GEMM A).
// LN3 additionally exports h (== gem row) at t==SEQ as the hn output.
template <bool HN>
__global__ __launch_bounds__(256) void ln_mid_kernel(
    const float* __restrict__ gem, const float* __restrict__ resIn,
    const float* __restrict__ g, const float* __restrict__ bta,
    float* __restrict__ outF, bf16_t* __restrict__ outB, float* __restrict__ hn_out) {
  __shared__ float rbuf[12];
  int r = blockIdx.x, b = r / LSTEP, t = r % LSTEP;
  int tid = threadIdx.x;
  size_t base = (size_t)r * 512;
  float g0 = gem[base + tid], g1 = gem[base + tid + 256];
  if (HN && t == SEQ) {
    hn_out[(size_t)b * 512 + tid] = g0;
    hn_out[(size_t)b * 512 + tid + 256] = g1;
  }
  float x0 = g0 + resIn[base + tid];
  float x1 = g1 + resIn[base + tid + 256];
  float mean = block_reduce_sum(x0 + x1, rbuf, tid, 8) * (1.0f / 512.0f);
  float d0 = x0 - mean, d1 = x1 - mean;
  float var = block_reduce_sum(d0 * d0 + d1 * d1, rbuf, tid, 9) * (1.0f / 512.0f);
  float rstd = rsqrtf(var + LN_EPS);
  float y0 = d0 * rstd * g[tid] + bta[tid];
  float y1 = d1 * rstd * g[tid + 256] + bta[tid + 256];
  outF[base + tid] = y0;
  outF[base + tid + 256] = y1;
  outB[base + tid] = (bf16_t)y0;
  outB[base + tid + 256] = (bf16_t)y1;
}

// LN4: final LN, written straight to d_out (drop t==0 rows).
__global__ __launch_bounds__(256) void ln4_kernel(
    const float* __restrict__ gem, const float* __restrict__ resIn,
    const float* __restrict__ g, const float* __restrict__ bta,
    float* __restrict__ out) {
  __shared__ float rbuf[12];
  int r = blockIdx.x, b = r / LSTEP, t = r % LSTEP;
  int tid = threadIdx.x;
  size_t base = (size_t)r * 512;
  float x0 = gem[base + tid] + resIn[base + tid];
  float x1 = gem[base + tid + 256] + resIn[base + tid + 256];
  float mean = block_reduce_sum(x0 + x1, rbuf, tid, 8) * (1.0f / 512.0f);
  float d0 = x0 - mean, d1 = x1 - mean;
  float var = block_reduce_sum(d0 * d0 + d1 * d1, rbuf, tid, 9) * (1.0f / 512.0f);
  float rstd = rsqrtf(var + LN_EPS);
  if (t >= 1) {
    size_t o = ((size_t)b * SEQ + (t - 1)) * 512;
    out[o + tid] = d0 * rstd * g[tid] + bta[tid];
    out[o + tid + 256] = d1 * rstd * g[tid + 256] + bta[tid + 256];
  }
}

// ---------------------------------------------------------------------------
// Host orchestration
// ---------------------------------------------------------------------------
static inline char* carve(char*& p, size_t bytes) {
  char* r = p;
  p += (bytes + 255) & ~(size_t)255;
  return r;
}

extern "C" void kernel_launch(void* const* d_in, const int* in_sizes, int n_in,
                              void* d_out, int out_size, void* d_ws, size_t ws_size,
                              hipStream_t stream) {
  const float* input = (const float*)d_in[0];
  const float* h0    = (const float*)d_in[2];
  const float* wq_c  = (const float*)d_in[3];
  const float* fc_c  = (const float*)d_in[5];
  const float* wq_g  = (const float*)d_in[6];
  const float* wk_g  = (const float*)d_in[7];
  const float* fc_g  = (const float*)d_in[8];
  const float* wq_x  = (const float*)d_in[9];
  const float* fc_x  = (const float*)d_in[11];
  const float* conv1 = (const float*)d_in[12];
  const float* conv2 = (const float*)d_in[13];
  const float* ln1_g = (const float*)d_in[14];
  const float* ln1_b = (const float*)d_in[15];
  const float* ln2_g = (const float*)d_in[16];
  const float* ln2_b = (const float*)d_in[17];
  const float* ln3_g = (const float*)d_in[18];
  const float* ln3_b = (const float*)d_in[19];
  const float* ln4_g = (const float*)d_in[20];
  const float* ln4_b = (const float*)d_in[21];
  const float* mlp_w1 = (const float*)d_in[22];
  const float* mlp_b1 = (const float*)d_in[23];
  const float* mlp_w2 = (const float*)d_in[24];
  const float* mlp_b2 = (const float*)d_in[25];
  float* out    = (float*)d_out;                           // [16,1024,512]
  float* hn_out = (float*)d_out + (size_t)BS * SEQ * 512;  // [16,512]

  char* p = (char*)d_ws;
  const size_t W = 512 * 512;
  bf16_t* wqcB = (bf16_t*)carve(p, W * 2);
  bf16_t* fccB = (bf16_t*)carve(p, W * 2);
  bf16_t* wqgB = (bf16_t*)carve(p, W * 2);
  bf16_t* wkgB = (bf16_t*)carve(p, W * 2);
  bf16_t* fcgB = (bf16_t*)carve(p, W * 2);
  bf16_t* wqxB = (bf16_t*)carve(p, W * 2);
  bf16_t* fcxB = (bf16_t*)carve(p, W * 2);
  bf16_t* w1B  = (bf16_t*)carve(p, (size_t)D_FF * 512 * 2);
  bf16_t* w2B  = (bf16_t*)carve(p, (size_t)512 * D_FF * 2);
  bf16_t* inB  = (bf16_t*)carve(p, (size_t)RK_ROWS * 512 * 2);
  bf16_t* KgB  = (bf16_t*)carve(p, (size_t)RK_ROWS * 512 * 2);
  bf16_t* VgT  = (bf16_t*)carve(p, (size_t)RK_ROWS * 512 * 2);
  bf16_t* actB0 = (bf16_t*)carve(p, (size_t)R_PAD * 512 * 2);
  bf16_t* actB1 = (bf16_t*)carve(p, (size_t)R_PAD * 512 * 2);
  bf16_t* uB    = (bf16_t*)carve(p, (size_t)R_ROWS * D_FF * 2);
  float* gemF  = (float*)carve(p, (size_t)R_ROWS * 512 * 4);
  float* resF0 = (float*)carve(p, (size_t)R_ROWS * 512 * 4);
  float* resF1 = (float*)carve(p, (size_t)R_ROWS * 512 * 4);
  (void)ws_size; (void)n_in; (void)in_sizes; (void)out_size;

  // f32 -> bf16 weight/activation conversions
  auto cvt = [&](const float* s, bf16_t* d, size_t n) {
    cvt_kernel<<<dim3((unsigned)((n + 255) / 256)), 256, 0, stream>>>(s, d, n);
  };
  cvt(wq_c, wqcB, W); cvt(fc_c, fccB, W); cvt(wq_g, wqgB, W); cvt(wk_g, wkgB, W);
  cvt(fc_g, fcgB, W); cvt(wq_x, wqxB, W); cvt(fc_x, fcxB, W);
  cvt(mlp_w1, w1B, (size_t)D_FF * 512); cvt(mlp_w2, w2B, (size_t)512 * D_FF);
  cvt(input, inB, (size_t)RK_ROWS * 512);

  const dim3 blk(256);
  const dim3 gR(R_ROWS / 16, 1);    // 1025 M-tiles over the 16400 rows
  const dim3 gK(RK_ROWS / 16, 1);   // 1024 M-tiles over input rows

  // Kg = input @ wk_g.T ; Vg = input @ wq_g.T (stored transposed per (b,h))
  gemm_kernel<EPI_BF16, 512><<<gK, blk, 0, stream>>>(inB, 512, wkgB, 512,
                                                     nullptr, KgB, 512, nullptr);
  gemm_kernel<EPI_VGT, 512><<<gK, blk, 0, stream>>>(inB, 512, wqgB, 512,
                                                    nullptr, VgT, 512, nullptr);

  // xc = conv1(X)
  conv1_kernel<<<dim3(R_ROWS), blk, 0, stream>>>(input, h0, conv1, actB0);
  // t1 = xc @ wq_c.T ; a1 = t1 @ fc_c.T
  gemm_kernel<EPI_BF16, 512><<<gR, blk, 0, stream>>>(actB0, 512, wqcB, 512,
                                                     nullptr, actB1, 512, nullptr);
  gemm_kernel<EPI_F32, 512><<<gR, blk, 0, stream>>>(actB1, 512, fccB, 512,
                                                    gemF, nullptr, 512, nullptr);
  // x1 = LN1(a1 + X) ; x2 = conv2(x1)
  ln1_kernel<<<dim3(R_ROWS), blk, 0, stream>>>(gemF, input, h0, ln1_g, ln1_b, conv2,
                                               resF0, actB0);
  // Q = x2 @ wq_g.T
  gemm_kernel<EPI_BF16, 512><<<gR, blk, 0, stream>>>(actB0, 512, wqgB, 512,
                                                     nullptr, actB1, 512, nullptr);
  // attention over the 1024 precomputed keys
  attn_kernel<<<dim3(17, BS * 8), dim3(128), 0, stream>>>(actB1, KgB, VgT, actB0);
  // g = att @ fc_g.T
  gemm_kernel<EPI_F32, 512><<<gR, blk, 0, stream>>>(actB0, 512, fcgB, 512,
                                                    gemF, nullptr, 512, nullptr);
  // x2n = LN2(g + x1)
  ln_mid_kernel<false><<<dim3(R_ROWS), blk, 0, stream>>>(gemF, resF0, ln2_g, ln2_b,
                                                         resF1, actB1, nullptr);
  // h = (x2n @ wq_x.T) @ fc_x.T
  gemm_kernel<EPI_BF16, 512><<<gR, blk, 0, stream>>>(actB1, 512, wqxB, 512,
                                                     nullptr, actB0, 512, nullptr);
  gemm_kernel<EPI_F32, 512><<<gR, blk, 0, stream>>>(actB0, 512, fcxB, 512,
                                                    gemF, nullptr, 512, nullptr);
  // x3 = LN3(h + x2n) ; export hn = h at t==SEQ
  ln_mid_kernel<true><<<dim3(R_ROWS), blk, 0, stream>>>(gemF, resF1, ln3_g, ln3_b,
                                                        resF0, actB1, hn_out);
  // u = gelu(x3 @ w1.T + b1) ; y = u @ w2.T + b2
  gemm_kernel<EPI_GELU_BF16, 512><<<dim3(R_ROWS / 16, 4), blk, 0, stream>>>(
      actB1, 512, w1B, 512, nullptr, uB, D_FF, mlp_b1);
  gemm_kernel<EPI_BIAS_F32, 2048><<<gR, blk, 0, stream>>>(uB, 2048, w2B, 2048,
                                                          gemF, nullptr, 512, mlp_b2);
  // y4 = LN4(y + x3) -> output rows (t >= 1)
  ln4_kernel<<<dim3(R_ROWS), blk, 0, stream>>>(gemF, resF0, ln4_g, ln4_b, out);
}
